// VariationalShuffle_88948772700688
// MI455X (gfx1250) — matrix operand order repaired and checked by reading, not compiled, not run on hardware
//
#include <hip/hip_runtime.h>
#include <hip/hip_bf16.h>

// ---------------------------------------------------------------------------
// VariationalShuffle on MI455X (gfx1250): all GEMMs via v_wmma_f32_16x16x32_bf16
// prep (weights->bf16 W^T, x->bf16) -> conv1 (edgeconv+leaky, dual fp32/bf16 h)
// -> conv2 (mean|logvar fused + reparam + point_shuffle, bf16 z)
// -> dec (z @ W_dec + b, pure-load A fragments).
// ---------------------------------------------------------------------------

typedef __attribute__((ext_vector_type(16))) __bf16 v16bf;
typedef __attribute__((ext_vector_type(8)))  float  v8f;

union FragU { unsigned u[8]; v16bf v; };

// HW packed convert: D[15:0]=bf16(lo), D[31:16]=bf16(hi). Probed via assembler.
__device__ __forceinline__ unsigned pk_bf16(float lo, float hi) {
  unsigned r;
  asm("v_cvt_pk_bf16_f32 %0, %1, %2" : "=v"(r) : "v"(lo), "v"(hi));
  return r;
}

__device__ __forceinline__ unsigned short bf16_1(float x) {
  return (unsigned short)(pk_bf16(x, x) & 0xFFFFu);
}

// ---------------------------------------------------------------------------
// Fragment builders. 16-bit A-matrix 16x32 layout (ISA 7.12.2):
// lane<16: M=lane, regs0-3 K=kb..kb+7, regs4-7 K=kb+16..kb+23
// lane>=16: same with kb += 8.  Each reg packs 2 consecutive K (bf16).
// ---------------------------------------------------------------------------

// A fragment straight from a bf16 row: two contiguous 16B loads, no VALU.
__device__ __forceinline__ v16bf a_frag_bf16(const unsigned short* __restrict__ rowp,
                                             int kb) {
  const uint4* p0 = reinterpret_cast<const uint4*>(rowp + kb);
  const uint4* p1 = reinterpret_cast<const uint4*>(rowp + kb + 16);
  uint4 q0 = p0[0], q1 = p1[0];
  FragU f;
  f.u[0] = q0.x; f.u[1] = q0.y; f.u[2] = q0.z; f.u[3] = q0.w;
  f.u[4] = q1.x; f.u[5] = q1.y; f.u[6] = q1.z; f.u[7] = q1.w;
  return f.v;
}

// A fragment = bf16(src_row - dst_row) from two fp32 rows (diff in fp32).
__device__ __forceinline__ v16bf a_frag_diff(const float* __restrict__ srcp,
                                             const float* __restrict__ dstp, int kb) {
  const float4* s0p = reinterpret_cast<const float4*>(srcp + kb);
  const float4* s1p = reinterpret_cast<const float4*>(srcp + kb + 16);
  const float4* d0p = reinterpret_cast<const float4*>(dstp + kb);
  const float4* d1p = reinterpret_cast<const float4*>(dstp + kb + 16);
  float4 sa0 = s0p[0], sa1 = s0p[1], sb0 = s1p[0], sb1 = s1p[1];
  float4 da0 = d0p[0], da1 = d0p[1], db0 = d1p[0], db1 = d1p[1];
  FragU f;
  f.u[0] = pk_bf16(sa0.x - da0.x, sa0.y - da0.y);
  f.u[1] = pk_bf16(sa0.z - da0.z, sa0.w - da0.w);
  f.u[2] = pk_bf16(sa1.x - da1.x, sa1.y - da1.y);
  f.u[3] = pk_bf16(sa1.z - da1.z, sa1.w - da1.w);
  f.u[4] = pk_bf16(sb0.x - db0.x, sb0.y - db0.y);
  f.u[5] = pk_bf16(sb0.z - db0.z, sb0.w - db0.w);
  f.u[6] = pk_bf16(sb1.x - db1.x, sb1.y - db1.y);
  f.u[7] = pk_bf16(sb1.z - db1.z, sb1.w - db1.w);
  return f.v;
}

// 16-bit B-matrix 32x16 fragment: regs0-3 K=kb..kb+7, regs4-7 K=kb+16..kb+23 (kb += 8 for lane>=16),
// col N = lane&15. Weights stored transposed [Cout][K2] bf16 -> 2 b128 loads.
__device__ __forceinline__ v16bf b_frag(const unsigned short* __restrict__ wt,
                                        int K2, int col, int kb) {
  const uint4* p0 = reinterpret_cast<const uint4*>(wt + col * K2 + kb);
  const uint4* p1 = reinterpret_cast<const uint4*>(wt + col * K2 + kb + 16);
  uint4 q0 = p0[0], q1 = p1[0];
  FragU f;
  f.u[0] = q0.x; f.u[1] = q0.y; f.u[2] = q0.z; f.u[3] = q0.w;
  f.u[4] = q1.x; f.u[5] = q1.y; f.u[6] = q1.z; f.u[7] = q1.w;
  return f.v;
}

__device__ __forceinline__ v8f wmma_bf16(v16bf a, v16bf b, v8f c) {
  return __builtin_amdgcn_wmma_f32_16x16x32_bf16(false, a, false, b, (short)0, c,
                                                 false, false);
}

__device__ __forceinline__ float max8(v8f a) {
  return fmaxf(fmaxf(fmaxf(a[0], a[1]), fmaxf(a[2], a[3])),
               fmaxf(fmaxf(a[4], a[5]), fmaxf(a[6], a[7])));
}

// ---------------------------------------------------------------------------
// prep: W_enc[128,64] -> W1T[64][128]bf16 ; {W_mean|W_logvar}[128,256] ->
//       W2T[512][128]bf16 ; W_dec[64,64] -> WDT[64][64]bf16
// ---------------------------------------------------------------------------
__global__ void k_prep(const float* __restrict__ We, const float* __restrict__ Wm,
                       const float* __restrict__ Wl, const float* __restrict__ Wd,
                       unsigned short* __restrict__ W1T,
                       unsigned short* __restrict__ W2T,
                       unsigned short* __restrict__ WDT) {
  int i = blockIdx.x * blockDim.x + threadIdx.x;
  if (i < 64 * 128) {
    int n = i >> 7, k = i & 127;
    W1T[n * 128 + k] = bf16_1(We[k * 64 + n]);
  } else if (i < 64 * 128 + 512 * 128) {
    int j = i - 64 * 128;
    int n = j >> 7, k = j & 127;
    float v = (n < 256) ? Wm[k * 256 + n] : Wl[k * 256 + (n - 256)];
    W2T[n * 128 + k] = bf16_1(v);
  } else if (i < 64 * 128 + 512 * 128 + 64 * 64) {
    int j = i - (64 * 128 + 512 * 128);
    int n = j >> 6, k = j & 63;
    WDT[n * 64 + k] = bf16_1(Wd[k * 64 + n]);
  }
}

// Row-major fp32 -> bf16 (pairs), for Xb.
__global__ void k_cvt(const float* __restrict__ src, unsigned* __restrict__ dst,
                      int nPairs) {
  int i = blockIdx.x * blockDim.x + threadIdx.x;
  if (i < nPairs) {
    const float2 v = reinterpret_cast<const float2*>(src)[i];
    dst[i] = pk_bf16(v.x, v.y);
  }
}

// ---------------------------------------------------------------------------
// conv1: h = leaky_relu( max_over_8_edges( [x_i, x_j-x_i] @ W_enc ) + b )
// One wave = 16 edge rows = 2 nodes. K=128 concat -> 4 wmma k-steps.
// x_i fragments load pre-converted bf16; diff fragments convert from fp32.
// ---------------------------------------------------------------------------
__global__ __launch_bounds__(256) void k_conv1(const float* __restrict__ X,
                                               const unsigned short* __restrict__ Xb,
                                               const int* __restrict__ esrc,
                                               const unsigned short* __restrict__ W1T,
                                               const float* __restrict__ be,
                                               float* __restrict__ H,
                                               unsigned short* __restrict__ Hb,
                                               int nPairs) {
  int lane = threadIdx.x & 31;
  int wid = (blockIdx.x * blockDim.x + threadIdx.x) >> 5;
  if (wid >= nPairs) return;
  int laneHi = lane >> 4;
  int row = lane & 15;                 // A-matrix row M this lane feeds
  int n0 = wid * 2;
  int node = n0 + (row >> 3);          // rows 0-7 -> n0, rows 8-15 -> n1
  int e = node * 8 + (row & 7);
  int s = esrc[e];
  const float* dstp = X + node * 64;
  const float* srcp = X + s * 64;
  const unsigned short* dstb = Xb + node * 64;

  v16bf af[4];
#pragma unroll
  for (int ks = 0; ks < 4; ++ks) {
    int kb = ks * 32 + laneHi * 8;     // concat-K in [0,128)
    af[ks] = (kb < 64) ? a_frag_bf16(dstb, kb) : a_frag_diff(srcp, dstp, kb - 64);
  }

  int colw = lane & 15;
#pragma unroll
  for (int nt = 0; nt < 4; ++nt) {
    v8f acc = {0.f, 0.f, 0.f, 0.f, 0.f, 0.f, 0.f, 0.f};
#pragma unroll
    for (int ks = 0; ks < 4; ++ks) {
      v16bf bf = b_frag(W1T, 128, nt * 16 + colw, ks * 32 + laneHi * 8);
      acc = wmma_bf16(af[ks], bf, acc);
    }
    // rows 0-7 (node n0) live in lanes 0-15 across regs 0-7; rows 8-15 in 16-31
    float m = max8(acc);
    int col = nt * 16 + colw;
    m += be[col];
    m = m > 0.f ? m : 0.2f * m;        // leaky_relu(0.2)
    int idx = (n0 + laneHi) * 64 + col;
    H[idx] = m;                        // fp32 (for exact diffs in conv2)
    Hb[idx] = bf16_1(m);               // bf16 (for x_i fragments in conv2)
  }
}

// ---------------------------------------------------------------------------
// conv2: mean (tiles 0..15) and logvar (tiles 16..31) share A fragments.
// Epilogue fuses reparametrize + point_shuffle, stores z as bf16.
// ---------------------------------------------------------------------------
__global__ __launch_bounds__(256) void k_conv2(const float* __restrict__ H,
                                               const unsigned short* __restrict__ Hb,
                                               const int* __restrict__ esrc,
                                               const unsigned short* __restrict__ W2T,
                                               const float* __restrict__ bm,
                                               const float* __restrict__ bl,
                                               const float* __restrict__ noise,
                                               unsigned short* __restrict__ Zb,
                                               int nPairs) {
  int lane = threadIdx.x & 31;
  int wid = (blockIdx.x * blockDim.x + threadIdx.x) >> 5;
  if (wid >= nPairs) return;
  int laneHi = lane >> 4;
  int row = lane & 15;
  int n0 = wid * 2;
  int node = n0 + (row >> 3);
  int e = node * 8 + (row & 7);
  int s = esrc[e];
  const float* dstp = H + node * 64;
  const float* srcp = H + s * 64;
  const unsigned short* dstb = Hb + node * 64;

  v16bf af[4];
#pragma unroll
  for (int ks = 0; ks < 4; ++ks) {
    int kb = ks * 32 + laneHi * 8;
    af[ks] = (kb < 64) ? a_frag_bf16(dstb, kb) : a_frag_diff(srcp, dstp, kb - 64);
  }

  int colw = lane & 15;
  float mval[16];
#pragma unroll
  for (int nt = 0; nt < 16; ++nt) {    // z_mean tiles (W2T rows 0..255)
    v8f acc = {0.f, 0.f, 0.f, 0.f, 0.f, 0.f, 0.f, 0.f};
#pragma unroll
    for (int ks = 0; ks < 4; ++ks) {
      v16bf bf = b_frag(W2T, 128, nt * 16 + colw, ks * 32 + laneHi * 8);
      acc = wmma_bf16(af[ks], bf, acc);
    }
    int f = nt * 16 + colw;
    mval[nt] = max8(acc) + bm[f];
  }
#pragma unroll
  for (int nt = 0; nt < 16; ++nt) {    // z_logvar tiles (W2T rows 256..511)
    v8f acc = {0.f, 0.f, 0.f, 0.f, 0.f, 0.f, 0.f, 0.f};
#pragma unroll
    for (int ks = 0; ks < 4; ++ks) {
      v16bf bf = b_frag(W2T, 128, (nt + 16) * 16 + colw, ks * 32 + laneHi * 8);
      acc = wmma_bf16(af[ks], bf, acc);
    }
    int f = nt * 16 + colw;
    float lv = max8(acc) + bl[f];
    int r = f & 3, c2 = f >> 2;                        // point_shuffle mapping
    int orow = (n0 + laneHi) * 4 + r;
    float z = mval[nt] + noise[orow * 64 + c2] * __expf(0.5f * lv);
    Zb[orow * 64 + c2] = bf16_1(z);    // decoder consumes bf16 anyway
  }
}

// ---------------------------------------------------------------------------
// dec: out = Zb @ W_dec + b_dec.  One wave = 16 rows; A fragments are pure loads.
// ---------------------------------------------------------------------------
__global__ __launch_bounds__(256) void k_dec(const unsigned short* __restrict__ Zb,
                                             const unsigned short* __restrict__ WDT,
                                             const float* __restrict__ bd,
                                             float* __restrict__ out, int nTiles) {
  int lane = threadIdx.x & 31;
  int wid = (blockIdx.x * blockDim.x + threadIdx.x) >> 5;
  if (wid >= nTiles) return;
  int laneHi = lane >> 4;
  int rowi = lane & 15;
  const unsigned short* zp = Zb + (wid * 16 + rowi) * 64;

  v16bf af[2];
#pragma unroll
  for (int ks = 0; ks < 2; ++ks) af[ks] = a_frag_bf16(zp, ks * 32 + laneHi * 8);

#pragma unroll
  for (int nt = 0; nt < 4; ++nt) {
    v8f acc = {0.f, 0.f, 0.f, 0.f, 0.f, 0.f, 0.f, 0.f};
#pragma unroll
    for (int ks = 0; ks < 2; ++ks) {
      v16bf bf = b_frag(WDT, 64, nt * 16 + rowi, ks * 32 + laneHi * 8);
      acc = wmma_bf16(af[ks], bf, acc);
    }
    int col = nt * 16 + rowi;
    float bb = bd[col];
#pragma unroll
    for (int r = 0; r < 8; ++r) {
      int orow = wid * 16 + r + laneHi * 8;            // C/D layout row mapping
      out[orow * 64 + col] = acc[r] + bb;
    }
  }
}

// ---------------------------------------------------------------------------
extern "C" void kernel_launch(void* const* d_in, const int* in_sizes, int n_in,
                              void* d_out, int out_size, void* d_ws, size_t ws_size,
                              hipStream_t stream) {
  constexpr int N = 50000, R = 4;
  const float* x       = (const float*)d_in[0];
  const int*   esrc    = (const int*)d_in[1];
  // d_in[2] = edge_dst (structured: repeat(arange(N),8)) -- not needed
  const float* noise   = (const float*)d_in[3];
  const float* W_enc   = (const float*)d_in[4];
  const float* b_enc   = (const float*)d_in[5];
  const float* W_mean  = (const float*)d_in[6];
  const float* b_mean  = (const float*)d_in[7];
  const float* W_logv  = (const float*)d_in[8];
  const float* b_logv  = (const float*)d_in[9];
  const float* W_dec   = (const float*)d_in[10];
  const float* b_dec   = (const float*)d_in[11];
  float* out = (float*)d_out;

  // workspace carve-up (bytes)
  char* ws = (char*)d_ws;
  size_t off = 0;
  unsigned short* W1T = (unsigned short*)(ws + off); off += 16384;
  unsigned short* W2T = (unsigned short*)(ws + off); off += 131072;
  unsigned short* WDT = (unsigned short*)(ws + off); off += 8192;
  unsigned short* Xb  = (unsigned short*)(ws + off); off += (size_t)N * 64 * 2;
  float*          H   = (float*)         (ws + off); off += (size_t)N * 64 * 4;
  unsigned short* Hb  = (unsigned short*)(ws + off); off += (size_t)N * 64 * 2;
  unsigned short* Zb  = (unsigned short*)(ws + off); // N*R*64*2 = 25.6 MB

  {
    int total = 64 * 128 + 512 * 128 + 64 * 64;
    k_prep<<<(total + 255) / 256, 256, 0, stream>>>(W_enc, W_mean, W_logv, W_dec,
                                                    W1T, W2T, WDT);
    int nPairsX = N * 64 / 2;
    k_cvt<<<(nPairsX + 255) / 256, 256, 0, stream>>>(x, (unsigned*)Xb, nPairsX);
  }
  {
    int nPairs = N / 2;                              // 25000 waves, 8 per block
    k_conv1<<<(nPairs * 32 + 255) / 256, 256, 0, stream>>>(x, Xb, esrc, W1T, b_enc,
                                                           H, Hb, nPairs);
    k_conv2<<<(nPairs * 32 + 255) / 256, 256, 0, stream>>>(H, Hb, esrc, W2T, b_mean,
                                                           b_logv, noise, Zb, nPairs);
  }
  {
    int nTiles = (N * R) / 16;                       // 12500 row-tiles
    k_dec<<<(nTiles * 32 + 255) / 256, 256, 0, stream>>>(Zb, WDT, b_dec, out,
                                                         nTiles);
  }
}
